// AttentionFreeTransformer_82325933129818
// MI455X (gfx1250) — matrix operand order repaired and checked by reading, not compiled, not run on hardware
//
#include <hip/hip_runtime.h>
#include <hip/hip_bf16.h>

// ---------------------------------------------------------------------------
// AFT block for MI455X (gfx1250, wave32).
// GEMMs: v_wmma_f32_16x16x32_f16, 8 WMMA per K-step per wave (32M x 64N tile),
// B tiles staged by the Tensor Data Mover (tensor_load_to_lds, double-buffered,
// TENSORcnt-waited).  Cumsum over T: 3-phase segmented scan.
// ---------------------------------------------------------------------------

typedef __attribute__((ext_vector_type(16))) _Float16 v16h;
typedef __attribute__((ext_vector_type(8)))  _Float16 v8h;
typedef __attribute__((ext_vector_type(8)))  float    v8f;
typedef __attribute__((ext_vector_type(4)))  uint32_t v4u;
typedef __attribute__((ext_vector_type(8)))  uint32_t v8u;

static constexpr int   Bc    = 4;
static constexpr int   Tc    = 8192;
static constexpr int   Dc    = 1024;      // n_embd == n_qkv
static constexpr int   BT    = Bc * Tc;   // 32768 tokens
static constexpr int   NSEG  = 32;
static constexpr int   SEGLEN= Tc / NSEG; // 256
static constexpr float EPS   = 1.1920929e-07f;

// LDS B tile: 32 K-rows x 128 N-cols of f16, row stride padded to 136 halves
// (TDM pad: 4 dwords every 64 dwords) -> stride 272B, 16B aligned, conflict-spread.
static constexpr int   BT_STRIDE = 136;               // halves per row
static constexpr int   BT_BUF_HALVES = 32 * BT_STRIDE;
static constexpr uint32_t BT_BUF_BYTES = BT_BUF_HALVES * 2;  // 8704

// ------------------------------------------------- TDM issue (inline asm)
// Loads a 2-D tile [32 rows x 128 cols] of f16 from wT[K][N] into LDS with
// row padding.  D# layout per CDNA5 ISA.
__device__ __forceinline__ void tdm_load_b_tile(const _Float16* gsrc,
                                                uint32_t lds_byte_off,
                                                uint32_t N, uint32_t K) {
    const uint64_t ga = (uint64_t)(uintptr_t)gsrc;
    v4u g0;
    g0[0] = 1u;                                            // count=1, user mode
    g0[1] = lds_byte_off;                                  // lds_addr
    g0[2] = (uint32_t)ga;                                  // global_addr[31:0]
    g0[3] = ((uint32_t)(ga >> 32) & 0x01FFFFFFu)           // global_addr[56:32]
          | (2u << 30);                                    // type=2 ("image")
    v8u g1;
    g1[0] = (1u << 16)        // data_size = 1 -> 2 bytes
          | (1u << 20)        // pad_enable
          | (5u << 22)        // pad_interval = 64 dwords (one 128-half row)
          | (3u << 25);       // pad_amount   = 4 dwords  (8 halves)
    g1[1] = (N & 0xFFFFu) << 16;                           // tensor_dim0[15:0]
    g1[2] = (N >> 16) | ((K & 0xFFFFu) << 16);             // dim0[31:16] | dim1[15:0]
    g1[3] = (K >> 16) | (128u << 16);                      // dim1[31:16] | tile_dim0=128
    g1[4] = 32u;                                           // tile_dim1=32, tile_dim2=0
    g1[5] = N;                                             // tensor_dim0_stride[31:0]
    g1[6] = 0u;                                            // stride[47:32] | dim1_stride lo
    g1[7] = 0u;
    asm volatile("tensor_load_to_lds %0, %1" :: "s"(g0), "s"(g1) : "memory");
}

// ---------------------------------------------------------------- converters
// w[N][K] f32 -> wT[K][N] f16 (32x32 LDS-tiled transpose)
__global__ void __launch_bounds__(256)
transpose_f32_to_f16_kernel(const float* __restrict__ w, _Float16* __restrict__ wT,
                            int N, int K) {
    __shared__ float tile[32][33];
    const int k0 = blockIdx.x * 32;
    const int n0 = blockIdx.y * 32;
    const int tx = threadIdx.x & 31;
    const int ty = (threadIdx.x >> 5) * 4;
    #pragma unroll
    for (int j = 0; j < 4; ++j)
        tile[ty + j][tx] = w[(size_t)(n0 + ty + j) * K + k0 + tx];
    __syncthreads();
    #pragma unroll
    for (int j = 0; j < 4; ++j)
        wT[(size_t)(k0 + ty + j) * N + n0 + tx] = (_Float16)tile[tx][ty + j];
}

// ------------------------------------------------------- rmsnorm(x) -> f16
__global__ void __launch_bounds__(256)
rmsnorm_x_kernel(const float* __restrict__ x, _Float16* __restrict__ xn) {
    __shared__ float red[256];
    const size_t base = (size_t)blockIdx.x * Dc + threadIdx.x * 4;
    float4 v = *(const float4*)(x + base);
    red[threadIdx.x] = v.x*v.x + v.y*v.y + v.z*v.z + v.w*v.w;
    __syncthreads();
    for (int off = 128; off > 0; off >>= 1) {
        if ((int)threadIdx.x < off) red[threadIdx.x] += red[threadIdx.x + off];
        __syncthreads();
    }
    const float r = rsqrtf(red[0] * (1.0f / Dc) + EPS);
    xn[base + 0] = (_Float16)(v.x * r);
    xn[base + 1] = (_Float16)(v.y * r);
    xn[base + 2] = (_Float16)(v.z * r);
    xn[base + 3] = (_Float16)(v.w * r);
}

// ------------------------------------------------------------ WMMA GEMM
// C[M,N] (f32) = A[M,K] (f16 row-major) * wT[K,N] (f16) [+ resid]
// Block: 256 threads = 8 waves; block tile 128(M) x 128(N).
// Wave tile 32(M) x 64(N): 2 A frags x 4 B frags = 8 WMMAs per 32-K step.
// B tile staged to LDS by TDM, double buffered, one barrier per step.
template <bool HAS_RESID>
__global__ void __launch_bounds__(256)
gemm_f16_wmma_kernel(const _Float16* __restrict__ A, const _Float16* __restrict__ BT16,
                     float* __restrict__ C, const float* __restrict__ resid,
                     int M, int N, int K) {
    __shared__ __align__(16) _Float16 Bt[2][BT_BUF_HALVES];

    const int lane = threadIdx.x & 31;
    const int wv   = threadIdx.x >> 5;
    const int wm   = wv >> 1;            // 0..3 -> 32-row subtiles
    const int wn   = wv & 1;             // 0..1 -> 64-col subtiles
    const int mBase = blockIdx.y * 128;
    const int nBase = blockIdx.x * 128;

    // A fragment addressing (16-bit A 16x32 layout, ISA 7.12.2)
    const int m0    = mBase + wm * 32 + (lane & 15);
    const int kHalf = (lane >> 4) * 8;           // 0 or 8
    const size_t arow0 = (size_t)m0 * K;
    const size_t arow1 = (size_t)(m0 + 16) * K;

    const uint32_t ldsBase = (uint32_t)(uintptr_t)(&Bt[0][0]);
    const bool wave0 = (threadIdx.x < 32);
    const int nk = K / 32;

    // prologue: stage first B tile
    if (wave0) {
        tdm_load_b_tile(BT16 + (size_t)nBase, ldsBase, (uint32_t)N, (uint32_t)K);
        __builtin_amdgcn_s_wait_tensorcnt(0);
    }
    __syncthreads();

    v8f acc0[4] = {};
    v8f acc1[4] = {};
    for (int i = 0; i < nk; ++i) {
        const int cur = i & 1;
        const int k0  = i * 32;
        if (i + 1 < nk && wave0)
            tdm_load_b_tile(BT16 + (size_t)(k0 + 32) * N + nBase,
                            ldsBase + (uint32_t)((i + 1) & 1) * BT_BUF_BYTES,
                            (uint32_t)N, (uint32_t)K);

        if (i + 1 < nk) {
            __builtin_prefetch(A + arow0 + k0 + 32 + kHalf, 0, 0);  // global_prefetch_b8
            __builtin_prefetch(A + arow1 + k0 + 32 + kHalf, 0, 0);
        }

        // A fragments (2x): rows m0 and m0+16
        v8h a0lo = *(const v8h*)(A + arow0 + k0 + kHalf);
        v8h a0hi = *(const v8h*)(A + arow0 + k0 + 16 + kHalf);
        v8h a1lo = *(const v8h*)(A + arow1 + k0 + kHalf);
        v8h a1hi = *(const v8h*)(A + arow1 + k0 + 16 + kHalf);
        v16h a0 = __builtin_shufflevector(a0lo, a0hi,
                  0,1,2,3,4,5,6,7,8,9,10,11,12,13,14,15);
        v16h a1 = __builtin_shufflevector(a1lo, a1hi,
                  0,1,2,3,4,5,6,7,8,9,10,11,12,13,14,15);

        // B fragments (4x) into distinct registers so WMMAs issue back-to-back
        const _Float16* bcol = &Bt[cur][wn * 64 + (lane & 15)];
        v16h b[4];
        #pragma unroll
        for (int f = 0; f < 4; ++f) {
            #pragma unroll
            for (int s = 0; s < 16; ++s) {
                const int kk = (s < 8) ? (kHalf + s) : (16 + kHalf + (s - 8));
                b[f][s] = bcol[kk * BT_STRIDE + f * 16];
            }
        }

        #pragma unroll
        for (int f = 0; f < 4; ++f)
            acc0[f] = __builtin_amdgcn_wmma_f32_16x16x32_f16(
                          false, a0, false, b[f], (short)0, acc0[f], false, false);
        #pragma unroll
        for (int f = 0; f < 4; ++f)
            acc1[f] = __builtin_amdgcn_wmma_f32_16x16x32_f16(
                          false, a1, false, b[f], (short)0, acc1[f], false, false);

        if (i + 1 < nk) {
            if (wave0) __builtin_amdgcn_s_wait_tensorcnt(0);
            __syncthreads();
        }
    }

    const int rbase = mBase + wm * 32 + 8 * (lane >> 4);
    #pragma unroll
    for (int f = 0; f < 4; ++f) {
        const int col = nBase + wn * 64 + f * 16 + (lane & 15);
        #pragma unroll
        for (int v = 0; v < 8; ++v) {
            const size_t o0 = (size_t)(rbase + v) * N + col;
            const size_t o1 = (size_t)(rbase + 16 + v) * N + col;
            float v0 = acc0[f][v];
            float v1 = acc1[f][v];
            if (HAS_RESID) { v0 += resid[o0]; v1 += resid[o1]; }
            C[o0] = v0;
            C[o1] = v1;
        }
    }
}

// -------------------- per-token: rmsnorm(q), rmsnorm(k), w=exp(k), kv=w*v
__global__ void __launch_bounds__(256)
qkv_post_kernel(const float* __restrict__ qkv, float* __restrict__ qn,
                float* __restrict__ wbuf, float* __restrict__ kvbuf) {
    __shared__ float rq_s[256], rk_s[256];
    const float* p = qkv + (size_t)blockIdx.x * (3 * Dc);
    const int c4 = threadIdx.x * 4;
    float4 q  = *(const float4*)(p + c4);
    float4 k  = *(const float4*)(p + Dc + c4);
    float4 vv = *(const float4*)(p + 2 * Dc + c4);
    rq_s[threadIdx.x] = q.x*q.x + q.y*q.y + q.z*q.z + q.w*q.w;
    rk_s[threadIdx.x] = k.x*k.x + k.y*k.y + k.z*k.z + k.w*k.w;
    __syncthreads();
    for (int off = 128; off > 0; off >>= 1) {
        if ((int)threadIdx.x < off) {
            rq_s[threadIdx.x] += rq_s[threadIdx.x + off];
            rk_s[threadIdx.x] += rk_s[threadIdx.x + off];
        }
        __syncthreads();
    }
    const float rq = rsqrtf(rq_s[0] * (1.0f / Dc) + EPS);
    const float rk = rsqrtf(rk_s[0] * (1.0f / Dc) + EPS);
    const size_t ob = (size_t)blockIdx.x * Dc + c4;
    float wq[4]  = {q.x, q.y, q.z, q.w};
    float wk[4]  = {k.x, k.y, k.z, k.w};
    float wvv[4] = {vv.x, vv.y, vv.z, vv.w};
    #pragma unroll
    for (int j = 0; j < 4; ++j) {
        const float we = __expf(wk[j] * rk);
        qn[ob + j]    = wq[j] * rq;
        wbuf[ob + j]  = we;
        kvbuf[ob + j] = we * wvv[j];
    }
}

// -------------------------------------------- scan phase 1: segment sums
__global__ void __launch_bounds__(256)
scan_segsum_kernel(const float* __restrict__ wbuf, const float* __restrict__ kvbuf,
                   float* __restrict__ segw, float* __restrict__ segkv) {
    const int gw   = blockIdx.x * 8 + (threadIdx.x >> 5);   // 0..4095
    const int lane = threadIdx.x & 31;
    const int seg  = gw & (NSEG - 1);
    const int grp  = gw >> 5;            // 0..127
    const int b    = grp >> 5;           // 0..3
    const int d    = (grp & 31) * 32 + lane;
    size_t idx = ((size_t)b * Tc + seg * SEGLEN) * Dc + d;
    float sw = 0.f, sk = 0.f;
    for (int t = 0; t < SEGLEN; ++t) {
        sw += wbuf[idx];
        sk += kvbuf[idx];
        idx += Dc;
    }
    const size_t so = ((size_t)(b * Dc + d)) * NSEG + seg;
    segw[so]  = sw;
    segkv[so] = sk;
}

// ------------------------------- scan phase 2: exclusive scan of segment sums
__global__ void __launch_bounds__(256)
scan_excl_kernel(float* __restrict__ segw, float* __restrict__ segkv) {
    const int row = blockIdx.x * 256 + threadIdx.x;   // 0..4095 = b*Dc+d
    const size_t base = (size_t)row * NSEG;
    float cw = 0.f, ck = 0.f;
    for (int s = 0; s < NSEG; ++s) {
        const float tw = segw[base + s];
        const float tk = segkv[base + s];
        segw[base + s]  = cw;
        segkv[base + s] = ck;
        cw += tw;
        ck += tk;
    }
}

// ------------------ scan phase 3: apply prefix, produce y = sig(q)*kv/(w+eps)
__global__ void __launch_bounds__(256)
scan_apply_kernel(const float* __restrict__ wbuf, const float* __restrict__ kvbuf,
                  const float* __restrict__ qn,
                  const float* __restrict__ segw, const float* __restrict__ segkv,
                  _Float16* __restrict__ y16) {
    const int gw   = blockIdx.x * 8 + (threadIdx.x >> 5);
    const int lane = threadIdx.x & 31;
    const int seg  = gw & (NSEG - 1);
    const int grp  = gw >> 5;
    const int b    = grp >> 5;
    const int d    = (grp & 31) * 32 + lane;
    const size_t so = ((size_t)(b * Dc + d)) * NSEG + seg;
    float cw = segw[so];
    float ck = segkv[so];
    size_t idx = ((size_t)b * Tc + seg * SEGLEN) * Dc + d;
    for (int t = 0; t < SEGLEN; ++t) {
        cw += wbuf[idx];
        ck += kvbuf[idx];
        const float qv  = qn[idx];
        const float sig = 1.0f / (1.0f + __expf(-qv));
        y16[idx] = (_Float16)(sig * (ck / (cw + 1e-6f)));
        idx += Dc;
    }
}

// -------------------------------------------------- swiglu: h = u * silu(g)
__global__ void __launch_bounds__(256)
swiglu_kernel(const float* __restrict__ uv, _Float16* __restrict__ h16) {
    const size_t i = (size_t)blockIdx.x * 256 + threadIdx.x;   // over BT*Dc
    const size_t row = i >> 10;
    const int    c   = (int)(i & 1023);
    const float u = uv[row * (2 * Dc) + c];
    const float g = uv[row * (2 * Dc) + Dc + c];
    h16[i] = (_Float16)(u * g / (1.0f + __expf(-g)));
}

// ---------------------------------------------------------------------------
extern "C" void kernel_launch(void* const* d_in, const int* in_sizes, int n_in,
                              void* d_out, int out_size, void* d_ws, size_t ws_size,
                              hipStream_t stream) {
    const float* x        = (const float*)d_in[0];   // [B,T,D]
    const float* w_qkv    = (const float*)d_in[1];   // [3D, D]
    const float* w_swiglu = (const float*)d_in[2];   // [2D, D]
    const float* w_out    = (const float*)d_in[3];   // [D, D]
    float* out = (float*)d_out;
    (void)in_sizes; (void)n_in; (void)out_size; (void)ws_size;

    char* ws = (char*)d_ws;
    size_t off = 0;
    auto alloc = [&](size_t bytes) { char* p = ws + off; off += (bytes + 255) & ~(size_t)255; return p; };

    _Float16* wqT16  = (_Float16*)alloc((size_t)3 * Dc * Dc * 2);  // [K][3D]
    _Float16* wswT16 = (_Float16*)alloc((size_t)2 * Dc * Dc * 2);  // [K][2D]
    _Float16* woT16  = (_Float16*)alloc((size_t)Dc * Dc * 2);      // [K][D]
    _Float16* xn16   = (_Float16*)alloc((size_t)BT * Dc * 2);      // reused as y16
    float*    qnb    = (float*)   alloc((size_t)BT * Dc * 4);
    float*    wb     = (float*)   alloc((size_t)BT * Dc * 4);      // reused as h16
    float*    kvb    = (float*)   alloc((size_t)BT * Dc * 4);
    float*    segw   = (float*)   alloc((size_t)Bc * Dc * NSEG * 4);
    float*    segkv  = (float*)   alloc((size_t)Bc * Dc * NSEG * 4);
    float*    qkv    = (float*)   alloc((size_t)BT * 3 * Dc * 4);  // reused as uv

    _Float16* y16 = xn16;            // xn dead after qkv GEMM
    float*    uv  = qkv;             // qkv dead after qkv_post
    _Float16* h16 = (_Float16*)wb;   // w dead after scan_apply

    // 1) weights -> f16, transposed to [K][N]
    transpose_f32_to_f16_kernel<<<dim3(Dc/32, 3*Dc/32), 256, 0, stream>>>(w_qkv,    wqT16,  3*Dc, Dc);
    transpose_f32_to_f16_kernel<<<dim3(Dc/32, 2*Dc/32), 256, 0, stream>>>(w_swiglu, wswT16, 2*Dc, Dc);
    transpose_f32_to_f16_kernel<<<dim3(Dc/32,   Dc/32), 256, 0, stream>>>(w_out,    woT16,    Dc, Dc);

    // 2) xn = rmsnorm(x) -> f16
    rmsnorm_x_kernel<<<BT, 256, 0, stream>>>(x, xn16);

    // 3) qkv = xn @ w_qkv^T
    gemm_f16_wmma_kernel<false><<<dim3(3 * Dc / 128, BT / 128), 256, 0, stream>>>(
        xn16, wqT16, qkv, nullptr, BT, 3 * Dc, Dc);

    // 4) q/k rmsnorm, w = exp(k), kv = w*v
    qkv_post_kernel<<<BT, 256, 0, stream>>>(qkv, qnb, wb, kvb);

    // 5) segmented cumsum over T, produce y -> f16
    scan_segsum_kernel<<<512, 256, 0, stream>>>(wb, kvb, segw, segkv);
    scan_excl_kernel<<<Bc * Dc / 256, 256, 0, stream>>>(segw, segkv);
    scan_apply_kernel<<<512, 256, 0, stream>>>(wb, kvb, qnb, segw, segkv, y16);

    // 6) uv = y @ w_swiglu^T
    gemm_f16_wmma_kernel<false><<<dim3(2 * Dc / 128, BT / 128), 256, 0, stream>>>(
        y16, wswT16, uv, nullptr, BT, 2 * Dc, Dc);

    // 7) h = u * silu(g) -> f16
    swiglu_kernel<<<(size_t)BT * Dc / 256, 256, 0, stream>>>(uv, h16);

    // 8) out = x + h @ w_out^T
    gemm_f16_wmma_kernel<true><<<dim3(Dc / 128, BT / 128), 256, 0, stream>>>(
        h16, woT16, out, x, BT, Dc, Dc);
}
